// SimpleRNN_35081292874104
// MI455X (gfx1250) — matrix operand order repaired and checked by reading, compiled, not run on hardware
//
#include <hip/hip_runtime.h>
#include <hip/hip_bf16.h>
#include <math.h>

// Problem constants (SimpleRNN reference)
constexpr int kB = 4;
constexpr int kT = 1024;
constexpr int kV = 32000;
constexpr int kH = 512;
constexpr int kM = kB * kT;  // 4096 output rows of the logits GEMM

// ---- WMMA vector types (CDNA5 / gfx1250, wave32) ----
typedef __attribute__((ext_vector_type(16))) _Float16 v16h;
typedef __attribute__((ext_vector_type(8)))  float    v8f;

union FragH {
    v16h h;
    uint4 u[2];
};

// --------------------------------------------------------------------------
// Kernel 1: recover token index from one-hot rows. One block per (b,t) row.
// Pure-bandwidth scan of the 524 MB one-hot tensor; float4 loads.
// --------------------------------------------------------------------------
__global__ __launch_bounds__(256)
void k_onehot_argmax(const float* __restrict__ X, int* __restrict__ idx) {
    const size_t row = blockIdx.x;                 // row = b*T + t
    const float4* xr4 = (const float4*)(X + row * (size_t)kV);
    for (int i = threadIdx.x; i < kV / 4; i += 256) {
        const float4 v = xr4[i];
        if (v.x > 0.5f) idx[row] = 4 * i;
        else if (v.y > 0.5f) idx[row] = 4 * i + 1;
        else if (v.z > 0.5f) idx[row] = 4 * i + 2;
        else if (v.w > 0.5f) idx[row] = 4 * i + 3;
    }
}

// --------------------------------------------------------------------------
// Kernel 2: xproj[t][b][:] = W_xh[idx[b][t]][:] + b_h   (fp32, [T,4,H])
// --------------------------------------------------------------------------
__global__ __launch_bounds__(256)
void k_gather_xproj(const float* __restrict__ Wxh, const float* __restrict__ bh,
                    const int* __restrict__ idx, float* __restrict__ xp) {
    const int bid = blockIdx.x;          // bid = t*4 + b
    const int t = bid >> 2;
    const int b = bid & 3;
    const int row = idx[b * kT + t];
    const float4* src = (const float4*)(Wxh + (size_t)row * kH);
    const float4* bh4 = (const float4*)bh;
    float4* dst = (float4*)(xp + (size_t)bid * kH);
    for (int h = threadIdx.x; h < kH / 4; h += 256) {
        const float4 w = src[h];
        const float4 c = bh4[h];
        dst[h] = make_float4(w.x + c.x, w.y + c.y, w.z + c.z, w.w + c.w);
    }
}

// --------------------------------------------------------------------------
// Kernel 3: transpose + convert fp32 [rows, cols] -> f16 [cols, rows].
// Coalesced both directions via a 32x33 LDS tile.
// --------------------------------------------------------------------------
__global__ __launch_bounds__(256)
void k_transpose_f16(const float* __restrict__ src, _Float16* __restrict__ dst,
                     int rows, int cols) {
    __shared__ float tile[32][33];
    const int n0 = blockIdx.x * 32;      // col tile in src
    const int k0 = blockIdx.y * 32;      // row tile in src
    const int tx = threadIdx.x & 31;
    const int ty = threadIdx.x >> 5;     // 0..7
#pragma unroll
    for (int r = 0; r < 4; ++r)
        tile[ty + 8 * r][tx] = src[(size_t)(k0 + ty + 8 * r) * cols + (n0 + tx)];
    __syncthreads();
#pragma unroll
    for (int r = 0; r < 4; ++r)
        dst[(size_t)(n0 + ty + 8 * r) * rows + (k0 + tx)] =
            (_Float16)tile[tx][ty + 8 * r];
}

// --------------------------------------------------------------------------
// Kernel 4: sequential RNN scan. ONE workgroup of 1024 threads (32 waves),
// one 16x16 N-tile per wave, K=512 per step.
//
// Register/LDS budget engineered against the wave32 VGPR cap at 8 waves/SIMD:
//   - B fragments for K=0..255 preloaded into 64 VGPRs per wave (8 ksteps).
//   - B fragments for K=256..511 parked in LDS: 512x256 f16 = 256 KB.
//   - h state double-buffered in LDS as f16: 2x16x512 = 32 KB.
//   Total LDS = 288 KB <= 320 KB/WGP (dynamic shared memory).
// Inner loop: 4x ds_load_b128 (A) + 2x ds_load_b128 (B-half) + 2 WMMA per
// 64-wide K chunk; NO global or scratch traffic on the critical path.
// Dual accumulators halve the dependent-WMMA chain.
// --------------------------------------------------------------------------
__global__ __launch_bounds__(1024)
void k_rnn_scan(const float* __restrict__ h0,        // [4,512] fp32
                const float* __restrict__ xp,        // [T,4,512] fp32
                const _Float16* __restrict__ whhT,   // [512(N),512(K)] f16
                _Float16* __restrict__ hs,           // [4096,512] f16, row=b*T+t
                float* __restrict__ hT) {            // [4,512] fp32
    extern __shared__ _Float16 smem[];
    _Float16* bl   = smem;                 // [512][256] f16 : W_hh^T K=256..511
    _Float16* hbuf = smem + 512 * 256;     // [2][16*512] f16 : h double buffer

    const int tid = threadIdx.x;

    // Stage W_hh^T columns K=256..511 into LDS (uint4 = 8 halfs per copy).
    for (int i = tid; i < (512 * 256) / 8; i += 1024) {
        const int n = i >> 5;              // 32 uint4 chunks per row
        const int c = i & 31;
        *(uint4*)(bl + n * 256 + c * 8) =
            *(const uint4*)(whhT + (size_t)n * kH + 256 + c * 8);
    }
    // Zero h buffers (padding rows 4..15 stay zero forever), then load h0.
    for (int i = tid; i < 2 * 16 * kH; i += 1024) hbuf[i] = (_Float16)0.0f;
    __syncthreads();
    for (int i = tid; i < kB * kH; i += 1024) hbuf[i] = (_Float16)h0[i];
    __syncthreads();

    const int lane = tid & 31;
    const int wave = tid >> 5;          // 0..31 -> N-tile
    const int lmod = lane & 15;
    const int lhi  = lane >> 4;         // 0/1
    const int n    = wave * 16 + lmod;  // output column (hidden index)
    const int mrow = lmod;              // A fragment row (= lane%16, all lanes)

    // Preload B fragments for K=0..255 into registers: 8 ksteps x 8 VGPRs.
    FragH breg[8];
#pragma unroll
    for (int j = 0; j < 8; ++j) {
        const _Float16* p = whhT + (size_t)n * kH + j * 32 + lhi * 16;
        breg[j].u[0] = *(const uint4*)(p);
        breg[j].u[1] = *(const uint4*)(p + 8);
    }

    const unsigned bl_base = (unsigned)(n * 256);   // halfs offset into bl

    for (int t = 0; t < kT; ++t) {
        const _Float16* hcur = hbuf + (t & 1) * (16 * kH);

        // Prefetch xp for this step early (independent of the WMMA chain).
        float xpv[4];
        if (lhi == 0) {
#pragma unroll
            for (int r = 0; r < 4; ++r)
                xpv[r] = xp[((size_t)t * 4 + r) * kH + n];
        }

        v8f acc0 = {}, acc1 = {};
        unsigned blo = bl_base;
        asm volatile("" : "+v"(blo));   // keep LDS-B loads inside the t-loop
#pragma unroll
        for (int j = 0; j < 8; ++j) {
            // K chunk j*32 .. j*32+31 : B register-resident.
            const int kA = j * 32 + lhi * 16;
            FragH a;
            a.u[0] = *(const uint4*)(hcur + mrow * kH + kA);
            a.u[1] = *(const uint4*)(hcur + mrow * kH + kA + 8);
            acc0 = __builtin_amdgcn_wmma_f32_16x16x32_f16(
                false, a.h, false, breg[j].h, (short)0, acc0, false, false);

            // K chunk 256+j*32 .. : B LDS-resident.
            const int kA2 = 256 + j * 32 + lhi * 16;
            FragH a2, b2;
            a2.u[0] = *(const uint4*)(hcur + mrow * kH + kA2);
            a2.u[1] = *(const uint4*)(hcur + mrow * kH + kA2 + 8);
            const _Float16* bp = bl + blo + j * 32 + lhi * 16;
            b2.u[0] = *(const uint4*)(bp);
            b2.u[1] = *(const uint4*)(bp + 8);
            acc1 = __builtin_amdgcn_wmma_f32_16x16x32_f16(
                false, a2.h, false, b2.h, (short)0, acc1, false, false);
        }

        // Epilogue: C/D layout -> lane holds N=lane%16, rows M = r + 8*lhi.
        // Only batch rows 0..3 are live (lanes 0..15, r<4).
        _Float16* hnext = hbuf + ((t + 1) & 1) * (16 * kH);
        if (lhi == 0) {
#pragma unroll
            for (int r = 0; r < 4; ++r) {
                const float v = tanhf(acc0[r] + acc1[r] + xpv[r]);
                const _Float16 hv = (_Float16)v;
                hnext[r * kH + n] = hv;
                hs[((size_t)r * kT + t) * kH + n] = hv;
                if (t == kT - 1) hT[r * kH + n] = v;
            }
        }
        __syncthreads();
    }
}

// --------------------------------------------------------------------------
// Kernel 5: logits GEMM.  out[4096, 32000] = hs[4096,512] @ W_hy + b_y.
//   Block = 256 threads = 8 waves as 2(M) x 4(N); wave tile = 16M x 64N
//   (4 accumulators). K=512 fully in registers; A/B fragments are direct
//   global b128 loads (W_hy^T f16 = 32.75 MB -> resident in 192 MB L2).
//   This kernel is HBM-write-bound (524 MB of logits).
// --------------------------------------------------------------------------
__global__ __launch_bounds__(256)
void k_gemm_logits(const _Float16* __restrict__ hs,    // [4096,512] f16
                   const _Float16* __restrict__ whyT,  // [32000,512] f16
                   const float* __restrict__ by,       // [32000]
                   float* __restrict__ out) {          // [4096,32000] fp32
    const int lane = threadIdx.x & 31;
    const int wave = threadIdx.x >> 5;     // 0..7
    const int wm = wave >> 2;              // 0..1
    const int wn = wave & 3;               // 0..3
    const int M0 = blockIdx.y * 32 + wm * 16;
    const int N0 = blockIdx.x * 256 + wn * 64;
    const int lmod = lane & 15;
    const int lhi  = lane >> 4;

    const _Float16* arow = hs + (size_t)(M0 + lmod) * kH;
    const _Float16* brow0 = whyT + (size_t)(N0 +  0 + lmod) * kH;
    const _Float16* brow1 = whyT + (size_t)(N0 + 16 + lmod) * kH;
    const _Float16* brow2 = whyT + (size_t)(N0 + 32 + lmod) * kH;
    const _Float16* brow3 = whyT + (size_t)(N0 + 48 + lmod) * kH;

    v8f acc0 = {}, acc1 = {}, acc2 = {}, acc3 = {};

#pragma unroll 4
    for (int kk = 0; kk < kH; kk += 32) {
        const int kA = kk + lhi * 16;
        const int kB_ = kk + lhi * 16;
        FragH a;
        a.u[0] = *(const uint4*)(arow + kA);
        a.u[1] = *(const uint4*)(arow + kA + 8);
        FragH b;
        b.u[0] = *(const uint4*)(brow0 + kB_);
        b.u[1] = *(const uint4*)(brow0 + kB_ + 8);
        acc0 = __builtin_amdgcn_wmma_f32_16x16x32_f16(
            false, a.h, false, b.h, (short)0, acc0, false, false);
        b.u[0] = *(const uint4*)(brow1 + kB_);
        b.u[1] = *(const uint4*)(brow1 + kB_ + 8);
        acc1 = __builtin_amdgcn_wmma_f32_16x16x32_f16(
            false, a.h, false, b.h, (short)0, acc1, false, false);
        b.u[0] = *(const uint4*)(brow2 + kB_);
        b.u[1] = *(const uint4*)(brow2 + kB_ + 8);
        acc2 = __builtin_amdgcn_wmma_f32_16x16x32_f16(
            false, a.h, false, b.h, (short)0, acc2, false, false);
        b.u[0] = *(const uint4*)(brow3 + kB_);
        b.u[1] = *(const uint4*)(brow3 + kB_ + 8);
        acc3 = __builtin_amdgcn_wmma_f32_16x16x32_f16(
            false, a.h, false, b.h, (short)0, acc3, false, false);
    }

    // Store + fused bias. Lane holds col = lane%16 within tile; rows r+8*lhi.
    v8f accs[4] = {acc0, acc1, acc2, acc3};
#pragma unroll
    for (int s = 0; s < 4; ++s) {
        const int col = N0 + s * 16 + lmod;
        const float bias = by[col];
#pragma unroll
        for (int r = 0; r < 8; ++r) {
            const int row = M0 + r + 8 * lhi;
            out[(size_t)row * kV + col] = accs[s][r] + bias;
        }
    }
}

// --------------------------------------------------------------------------
// Host launcher
// --------------------------------------------------------------------------
extern "C" void kernel_launch(void* const* d_in, const int* in_sizes, int n_in,
                              void* d_out, int out_size, void* d_ws, size_t ws_size,
                              hipStream_t stream) {
    (void)in_sizes; (void)n_in; (void)out_size; (void)ws_size;

    const float* X   = (const float*)d_in[0];  // [B,T,V] one-hot
    const float* h0  = (const float*)d_in[1];  // [B,H]
    const float* Wxh = (const float*)d_in[2];  // [V,H]
    const float* Whh = (const float*)d_in[3];  // [H,H]
    const float* bh  = (const float*)d_in[4];  // [H]
    const float* Why = (const float*)d_in[5];  // [H,V]
    const float* by  = (const float*)d_in[6];  // [V]

    float* logits = (float*)d_out;                       // [B,T,V]
    float* hT     = logits + (size_t)kB * kT * kV;       // [B,H]

    // Workspace layout (bytes), all 16B-aligned:
    //   idx   int   [4096]          @ 0          (16,384 B)
    //   xp    fp32  [T,4,H]         @ 16,384     (8,388,608 B)
    //   whhT  f16   [H,H]           @ 8,404,992  (524,288 B)
    //   whyT  f16   [V,H]           @ 8,929,280  (32,768,000 B)
    //   hs    f16   [4096,H]        @ 41,697,280 (4,194,304 B)  total ~45.9 MB
    char* ws = (char*)d_ws;
    int*      idx  = (int*)(ws + 0);
    float*    xp   = (float*)(ws + 16384);
    _Float16* whhT = (_Float16*)(ws + 8404992);
    _Float16* whyT = (_Float16*)(ws + 8929280);
    _Float16* hs   = (_Float16*)(ws + 41697280);

    // 1) one-hot -> indices (single pass over 524 MB input, float4 loads)
    k_onehot_argmax<<<kB * kT, 256, 0, stream>>>(X, idx);
    // 2) xproj gather + bias
    k_gather_xproj<<<kT * kB, 256, 0, stream>>>(Wxh, bh, idx, xp);
    // 3) weight transposes -> f16 (W_hh^T: [N,K]; W_hy^T: [V,K])
    k_transpose_f16<<<dim3(kH / 32, kH / 32), 256, 0, stream>>>(Whh, whhT, kH, kH);
    k_transpose_f16<<<dim3(kV / 32, kH / 32), 256, 0, stream>>>(Why, whyT, kH, kV);
    // 4) sequential scan on one WGP (32 waves); 288 KB dynamic LDS
    //    (256 KB W_hh^T K-half + 32 KB h double buffer)
    k_rnn_scan<<<1, 1024, 294912, stream>>>(h0, xp, whhT, hs, hT);
    // 5) big WMMA GEMM: 125 x 128 blocks, wave tile 16x64
    k_gemm_logits<<<dim3(kV / 256, kM / 32), 256, 0, stream>>>(hs, whyT, by, logits);
}
